// Gemma4AudioAttention_91199335563722
// MI455X (gfx1250) — matrix operand order repaired and checked by reading, not compiled, not run on hardware
//
#include <hip/hip_runtime.h>

// ---------------- constants ----------------
#define BSZ    16
#define SEQ    2400
#define HS     1024
#define NH     8
#define HD     128
#define CHK    12
#define CTX    24
#define PPOS   25            // CTX + 1 rel positions
#define NBLK   200           // SEQ / CHK
#define SOFTCAP 50.0f
#define Q_SCALEC (0.08838834764831845f / 0.6931471805599453f)   // D^-0.5 / ln2
#define K_SCALEC (1.3132616875182228f / 0.6931471805599453f)    // ln(1+e) / ln2

typedef unsigned short u16;
typedef __attribute__((ext_vector_type(16))) __bf16 v16bf;
typedef __attribute__((ext_vector_type(8)))  float  v8f;

union Frag { v16bf v; uint4 q[2]; };
union Acc  { v8f  v; float f[8]; };

__device__ __forceinline__ u16 f2bf(float f) {
    unsigned u = __float_as_uint(f);
    u += 0x7FFFu + ((u >> 16) & 1u);   // round-to-nearest-even
    return (u16)(u >> 16);
}

__device__ __forceinline__ void zero_acc(Acc& a) {
#pragma unroll
    for (int i = 0; i < 8; i++) a.f[i] = 0.0f;
}

// ---- CDNA5 async global->LDS copy (ASYNCcnt-tracked DMA, 16B per lane) ----
__device__ __forceinline__ void async_ld_b128(unsigned lds_off, const u16* gptr) {
    asm volatile("global_load_async_to_lds_b128 %0, %1, off"
                 : : "v"(lds_off), "v"(gptr) : "memory");
}
__device__ __forceinline__ void wait_async0() {
    asm volatile("s_wait_asynccnt 0x0" : : : "memory");
}

// ---------------- f32 -> bf16 convert ----------------
__global__ void cvt_f32_bf16(const float* __restrict__ in, u16* __restrict__ out, long n) {
    long i = (long)blockIdx.x * blockDim.x + threadIdx.x;
    long stride = (long)gridDim.x * blockDim.x;
    for (; i < n; i += stride) out[i] = f2bf(in[i]);
}

// ---------------- tiled WMMA GEMM: C[M,N] = A[M,K] * B[K,N]  (bf16 in) ----------------
// block = 256 threads (8 waves, 4(M) x 2(N)), tile 128x128, K step 32.
enum { EPI_BF16 = 0, EPI_Q = 1, EPI_K = 2, EPI_F32 = 3 };

template <int EPI>
__global__ __launch_bounds__(256) void gemm_bf16_wmma(
    const u16* __restrict__ A, const u16* __restrict__ Bw, void* __restrict__ Cout,
    int M, int N, int K, const float* __restrict__ pds)
{
    __shared__ __align__(16) u16 As[128 * 32];   // [m][k]
    __shared__ __align__(16) u16 Bs[128 * 32];   // transposed: [n][k]

    const int tid  = threadIdx.x;
    const int lane = tid & 31;
    const int wave = tid >> 5;
    const int wm   = wave & 3;      // wave row (32 M each)
    const int wn   = wave >> 2;     // wave col (64 N each)
    const int m0   = blockIdx.y * 128;
    const int n0   = blockIdx.x * 128;
    const int ln   = lane & 15;
    const int g    = lane >> 4;

    Acc acc[2][4];
#pragma unroll
    for (int i = 0; i < 2; i++)
#pragma unroll
        for (int j = 0; j < 4; j++) zero_acc(acc[i][j]);

    const int ar = tid >> 1;            // A tile row 0..127
    const int ak = (tid & 1) * 16;      // A tile k offset 0/16
    const int bk = tid >> 3;            // B tile k row 0..31
    const int bn = (tid & 7) * 16;      // B tile n offset 0..112

    // clamp out-of-range A rows (M=25 rel-GEMM): garbage rows feed only D rows
    // that are never stored (WMMA rows are independent)
    const int arow  = m0 + ar;
    const int asrc  = (arow < M) ? arow : (M - 1);
    const unsigned ldsA = (unsigned)(unsigned long long)(&As[ar * 32 + ak]);

    for (int k0 = 0; k0 < K; k0 += 32) {
        // ---- stage A via CDNA5 async DMA to LDS (no VGPR round-trip) ----
        const u16* gA = A + (size_t)asrc * K + k0 + ak;
        async_ld_b128(ldsA,      gA);
        async_ld_b128(ldsA + 16, gA + 8);

        // ---- stage B transposed (needs shuffle; VGPR-mediated) ----
        {
            const uint4* bp = reinterpret_cast<const uint4*>(Bw + (size_t)(k0 + bk) * N + n0 + bn);
            uint4 b0 = bp[0], b1 = bp[1];
            const u16* e0 = reinterpret_cast<const u16*>(&b0);
            const u16* e1 = reinterpret_cast<const u16*>(&b1);
#pragma unroll
            for (int e = 0; e < 8; e++) Bs[(bn + e) * 32 + bk] = e0[e];
#pragma unroll
            for (int e = 0; e < 8; e++) Bs[(bn + 8 + e) * 32 + bk] = e1[e];
        }
        wait_async0();       // our async writes to LDS are done
        __syncthreads();     // everyone's writes visible

        // prefetch next K tile (global_prefetch_b8)
        if (k0 + 32 < K) {
            __builtin_prefetch(A + (size_t)asrc * K + k0 + 32 + ak, 0, 0);
            __builtin_prefetch(Bw + (size_t)(k0 + 32 + bk) * N + n0 + bn, 0, 0);
        }

        // ---- fragments + WMMA ----
        Frag afr[2];
#pragma unroll
        for (int tm = 0; tm < 2; tm++) {
            const int row = wm * 32 + tm * 16 + ln;
            afr[tm].q[0] = *reinterpret_cast<const uint4*>(&As[row * 32 + 16 * g]);
            afr[tm].q[1] = *reinterpret_cast<const uint4*>(&As[row * 32 + 16 * g + 8]);
        }
#pragma unroll
        for (int tn = 0; tn < 4; tn++) {
            Frag bfr;
            const int col = wn * 64 + tn * 16 + ln;
            bfr.q[0] = *reinterpret_cast<const uint4*>(&Bs[col * 32 + 16 * g]);
            bfr.q[1] = *reinterpret_cast<const uint4*>(&Bs[col * 32 + 16 * g + 8]);
#pragma unroll
            for (int tm = 0; tm < 2; tm++) {
                acc[tm][tn].v = __builtin_amdgcn_wmma_f32_16x16x32_bf16(
                    false, afr[tm].v, false, bfr.v, (short)0, acc[tm][tn].v, false, false);
            }
        }
        __syncthreads();
    }

    // ---- epilogue ----
#pragma unroll
    for (int tm = 0; tm < 2; tm++) {
#pragma unroll
        for (int tn = 0; tn < 4; tn++) {
            const int colg = n0 + wn * 64 + tn * 16 + ln;
#pragma unroll
            for (int r = 0; r < 8; r++) {
                const int mg = m0 + wm * 32 + tm * 16 + g * 8 + r;
                if (mg < M) {
                    float v = acc[tm][tn].f[r];
                    if (EPI == EPI_Q) {
                        const float s = pds[colg & (HD - 1)];
                        v *= Q_SCALEC * log1pf(expf(s));      // softplus(per_dim_scale)
                    } else if (EPI == EPI_K) {
                        v *= K_SCALEC;
                    }
                    if (EPI == EPI_F32)
                        ((float*)Cout)[(size_t)mg * N + colg] = v;
                    else
                        ((u16*)Cout)[(size_t)mg * N + colg] = f2bf(v);
                }
            }
        }
    }
}

// ---------------- blocked relative attention: one wave32 per (b, block, head) ----------------
__global__ __launch_bounds__(32) void attn_kernel(
    const u16* __restrict__ qb, const u16* __restrict__ kb,
    const u16* __restrict__ vb, const u16* __restrict__ relb,
    u16* __restrict__ attb)
{
    __shared__ __align__(16) u16  vt[HD * 32];   // v^T: [d][key], keys padded to 32
    __shared__            float   sAC[16 * 32];  // content-content scores -> logits
    __shared__            float   sBD[16 * 32];  // content-position scores (pre-shift)
    __shared__ __align__(16) u16  sP[16 * 32];   // softmax probs, bf16, padded

    const int lane = threadIdx.x;
    const int h = blockIdx.x & (NH - 1);
    const int n = (blockIdx.x >> 3) % NBLK;
    const int b = blockIdx.x / (NH * NBLK);
    const int ln = lane & 15;
    const int g  = lane >> 4;

    // zero padded key columns 24..31 of v^T
    for (int idx = lane; idx < HD * 8; idx += 32) {
        const int d = idx >> 3, j = 24 + (idx & 7);
        vt[d * 32 + j] = 0;
    }
    // stage v context transposed (zeros for left padding p < 0)
    for (int idx = lane; idx < CTX * HD; idx += 32) {
        const int j = idx >> 7;          // key 0..23
        const int d = idx & (HD - 1);
        const int p = n * CHK - CHK + j; // PAST = CHK = 12
        u16 val = 0;
        if (p >= 0) val = vb[(size_t)(b * SEQ + p) * HS + h * HD + d];
        vt[d * 32 + j] = val;
    }
    __syncthreads();

    // ---- scores: ac = q k^T   bd = q rel^T   (K = 128 in 4 steps, N padded to 32) ----
    Acc ac[2], bd[2];
    zero_acc(ac[0]); zero_acc(ac[1]); zero_acc(bd[0]); zero_acc(bd[1]);

    int qpos = n * CHK + ln;             // rows 12..15 are don't-care (clamped, outputs unused)
    if (qpos > SEQ - 1) qpos = SEQ - 1;
    const u16* qptr = qb + (size_t)(b * SEQ + qpos) * HS + h * HD;

#pragma unroll
    for (int kk = 0; kk < 4; kk++) {
        Frag af;
        af.q[0] = *reinterpret_cast<const uint4*>(qptr + kk * 32 + 16 * g);
        af.q[1] = *reinterpret_cast<const uint4*>(qptr + kk * 32 + 16 * g + 8);
#pragma unroll
        for (int nt = 0; nt < 2; nt++) {
            Frag kf; kf.q[0] = uint4{0,0,0,0}; kf.q[1] = uint4{0,0,0,0};
            const int j = nt * 16 + ln;
            const int p = n * CHK - CHK + j;
            if (j < CTX && p >= 0) {
                const u16* kp = kb + (size_t)(b * SEQ + p) * HS + h * HD + kk * 32 + 16 * g;
                kf.q[0] = *reinterpret_cast<const uint4*>(kp);
                kf.q[1] = *reinterpret_cast<const uint4*>(kp + 8);
            }
            ac[nt].v = __builtin_amdgcn_wmma_f32_16x16x32_bf16(
                false, af.v, false, kf.v, (short)0, ac[nt].v, false, false);

            Frag rf; rf.q[0] = uint4{0,0,0,0}; rf.q[1] = uint4{0,0,0,0};
            const int pi = nt * 16 + ln;
            if (pi < PPOS) {
                const u16* rp = relb + (size_t)pi * HS + h * HD + kk * 32 + 16 * g;
                rf.q[0] = *reinterpret_cast<const uint4*>(rp);
                rf.q[1] = *reinterpret_cast<const uint4*>(rp + 8);
            }
            bd[nt].v = __builtin_amdgcn_wmma_f32_16x16x32_bf16(
                false, af.v, false, rf.v, (short)0, bd[nt].v, false, false);
        }
    }

    // spill score tiles to LDS (C layout: VGPR r -> row r + 8g, col = nt*16 + ln)
#pragma unroll
    for (int nt = 0; nt < 2; nt++)
#pragma unroll
        for (int r = 0; r < 8; r++) {
            sAC[(g * 8 + r) * 32 + nt * 16 + ln] = ac[nt].f[r];
            sBD[(g * 8 + r) * 32 + nt * 16 + ln] = bd[nt].f[r];
        }
    __syncthreads();

    // zero probs pad
    for (int i = lane; i < 16 * 32; i += 32) sP[i] = 0;

    // rel-shift (flat-index trick) + tanh softcap, in place into sAC
    for (int idx = lane; idx < CHK * CTX; idx += 32) {
        const int qq = idx / CTX, j = idx % CTX;
        const int f = qq * CTX + j;                       // flat index into [12,25]
        const float bdv = sBD[(f / PPOS) * 32 + (f % PPOS)];
        float lg = sAC[qq * 32 + j] + bdv;
        lg = tanhf(lg * (1.0f / SOFTCAP)) * SOFTCAP;
        sAC[qq * 32 + j] = lg;
    }
    __syncthreads();

    // softmax per query row (lanes 0..11)
    if (lane < CHK) {
        float mx = -1e30f;
        for (int j = 0; j < CTX; j++) mx = fmaxf(mx, sAC[lane * 32 + j]);
        float ex[CTX]; float sum = 0.0f;
        for (int j = 0; j < CTX; j++) { ex[j] = expf(sAC[lane * 32 + j] - mx); sum += ex[j]; }
        const float inv = 1.0f / sum;
        for (int j = 0; j < CTX; j++) sP[lane * 32 + j] = f2bf(ex[j] * inv);
    }
    __syncthreads();

    // ---- out = probs(16x32) x v^T tiles (32x16 each), D = 128 -> 8 WMMAs ----
    Frag pf;
    pf.q[0] = *reinterpret_cast<const uint4*>(&sP[ln * 32 + 16 * g]);
    pf.q[1] = *reinterpret_cast<const uint4*>(&sP[ln * 32 + 16 * g + 8]);
#pragma unroll
    for (int nt = 0; nt < 8; nt++) {
        Frag vf;
        const int d = nt * 16 + ln;
        vf.q[0] = *reinterpret_cast<const uint4*>(&vt[d * 32 + 16 * g]);
        vf.q[1] = *reinterpret_cast<const uint4*>(&vt[d * 32 + 16 * g + 8]);
        Acc o; zero_acc(o);
        o.v = __builtin_amdgcn_wmma_f32_16x16x32_bf16(
            false, pf.v, false, vf.v, (short)0, o.v, false, false);
#pragma unroll
        for (int r = 0; r < 8; r++) {
            const int m = g * 8 + r;
            if (m < CHK)
                attb[(size_t)(b * SEQ + n * CHK + m) * HS + h * HD + nt * 16 + ln] = f2bf(o.f[r]);
        }
    }
}

// ---------------- host-side orchestration ----------------
extern "C" void kernel_launch(void* const* d_in, const int* in_sizes, int n_in,
                              void* d_out, int out_size, void* d_ws, size_t ws_size,
                              hipStream_t stream) {
    const float* x     = (const float*)d_in[0];
    const float* pos   = (const float*)d_in[1];
    const float* Wq    = (const float*)d_in[2];
    const float* Wk    = (const float*)d_in[3];
    const float* Wv    = (const float*)d_in[4];
    const float* Wpost = (const float*)d_in[5];
    const float* Wrel  = (const float*)d_in[6];
    const float* pds   = (const float*)d_in[7];

    const long ROWS = (long)BSZ * SEQ;             // 38400
    const long XN   = ROWS * HS;                   // 39,321,600
    const long WN   = (long)HS * HS;               // 1,048,576
    const long PN   = (long)PPOS * HS;             // 25,600

    char* ws = (char*)d_ws;
    size_t off = 0;
    auto alloc = [&](size_t bytes) -> void* {
        void* p = ws + off;
        off += (bytes + 255) & ~(size_t)255;
        return p;
    };
    u16* xb     = (u16*)alloc(XN * 2);
    u16* qbuf   = (u16*)alloc(XN * 2);
    u16* kbuf   = (u16*)alloc(XN * 2);
    u16* vbuf   = (u16*)alloc(XN * 2);
    u16* attb   = (u16*)alloc(XN * 2);
    u16* Wqb    = (u16*)alloc(WN * 2);
    u16* Wkb    = (u16*)alloc(WN * 2);
    u16* Wvb    = (u16*)alloc(WN * 2);
    u16* Wpostb = (u16*)alloc(WN * 2);
    u16* Wrelb  = (u16*)alloc(WN * 2);
    u16* posb   = (u16*)alloc(PN * 2);
    u16* relb   = (u16*)alloc(PN * 2);

    // f32 -> bf16 conversions
    cvt_f32_bf16<<<4096, 256, 0, stream>>>(x, xb, XN);
    cvt_f32_bf16<<<1024, 256, 0, stream>>>(Wq, Wqb, WN);
    cvt_f32_bf16<<<1024, 256, 0, stream>>>(Wk, Wkb, WN);
    cvt_f32_bf16<<<1024, 256, 0, stream>>>(Wv, Wvb, WN);
    cvt_f32_bf16<<<1024, 256, 0, stream>>>(Wpost, Wpostb, WN);
    cvt_f32_bf16<<<1024, 256, 0, stream>>>(Wrel, Wrelb, WN);
    cvt_f32_bf16<<<100, 256, 0, stream>>>(pos, posb, PN);

    // projections (q with fused Q_SCALE*softplus, k with fused K_SCALE)
    dim3 gBig(HS / 128, (int)(ROWS / 128));        // 8 x 300
    gemm_bf16_wmma<EPI_Q><<<gBig, 256, 0, stream>>>(xb, Wqb, qbuf, (int)ROWS, HS, HS, pds);
    gemm_bf16_wmma<EPI_K><<<gBig, 256, 0, stream>>>(xb, Wkb, kbuf, (int)ROWS, HS, HS, nullptr);
    gemm_bf16_wmma<EPI_BF16><<<gBig, 256, 0, stream>>>(xb, Wvb, vbuf, (int)ROWS, HS, HS, nullptr);

    // rel_k = pos_emb @ Wrel  (M = 25)
    dim3 gRel(HS / 128, 1);
    gemm_bf16_wmma<EPI_BF16><<<gRel, 256, 0, stream>>>(posb, Wrelb, relb, PPOS, HS, HS, nullptr);

    // blocked relative attention
    attn_kernel<<<BSZ * NBLK * NH, 32, 0, stream>>>(qbuf, kbuf, vbuf, relb, attb);

    // final projection, f32 out
    gemm_bf16_wmma<EPI_F32><<<gBig, 256, 0, stream>>>(attb, Wpostb, d_out, (int)ROWS, HS, HS, nullptr);
}
